// MixedLoraLinear_733
// MI455X (gfx1250) — matrix operand order repaired and checked by reading, not compiled, not run on hardware
//
#include <hip/hip_runtime.h>
#include <hip/hip_bf16.h>

// MI455X / gfx1250 MixedLoraLinear:
//   out = x @ W^T + b + scaling[a_t] * ((x_t @ wa[a_t]) @ wb[a_t])
// Base GEMM in bf16 WMMA (f32 accumulate), LoRA rank-16 path in f32.

typedef __attribute__((ext_vector_type(16))) __bf16 v16bf;
typedef __attribute__((ext_vector_type(8)))  float  v8f;

#define T_TOK  8192
#define D_IN   4096
#define D_OUTF 4096
#define RANK   16
#define NADAPT 8

__device__ __forceinline__ unsigned short f2bf(float f) {
    return __builtin_bit_cast(unsigned short, (__bf16)f);   // RNE convert
}

// ---------------------------------------------------------------------------
// Kernel 1: H[t, 0:16] = scaling[a] * (x_t @ wa[a]),  tokid[t] = a
// Grid: (T/64, A). Segment s covers tokens [segment[s], segment[s+1]) with a
// uniform adapter id, so each block sees exactly one adapter.
// ---------------------------------------------------------------------------
__global__ __launch_bounds__(256) void lora_h_kernel(
    const float* __restrict__ x, const float* __restrict__ wa,
    const float* __restrict__ scaling, const int* __restrict__ segment,
    const int* __restrict__ lora_ids, float* __restrict__ H,
    int* __restrict__ tokid)
{
    const int s      = blockIdx.y;
    const int segEnd = segment[s + 1];
    const int t0     = segment[s] + blockIdx.x * 64;
    if (t0 >= segEnd) return;

    const int   a  = lora_ids[s];
    const float sc = scaling[a];
    const int tid  = threadIdx.x;
    const int r    = tid & 15;   // rank index 0..15
    const int g    = tid >> 4;   // token group 0..15 (4 tokens each)

    __shared__ __align__(16) float xsh[64][72];   // 64 tokens x 64 k (padded)
    __shared__ __align__(16) float wash[64][20];  // 64 k x 16 r (padded)

    float acc0 = 0.f, acc1 = 0.f, acc2 = 0.f, acc3 = 0.f;
    const float* wabase = wa + (size_t)a * D_IN * RANK;

    const int c4 = tid & 15, r0 = tid >> 4;   // x tile loaders
    const int rowW = tid >> 2, cW = tid & 3;  // wa tile loaders

    for (int k0 = 0; k0 < D_IN; k0 += 64) {
        // stage x: 64 tokens x 64 k  (coalesced float4)
        #pragma unroll
        for (int i = 0; i < 4; ++i) {
            const int row = r0 + 16 * i;
            const int t   = t0 + row;
            float4 v = make_float4(0.f, 0.f, 0.f, 0.f);
            if (t < segEnd)
                v = *(const float4*)(x + (size_t)t * D_IN + k0 + c4 * 4);
            *(float4*)&xsh[row][c4 * 4] = v;
        }
        // stage wa: 64 k x 16 r
        {
            float4 v = *(const float4*)(wabase + (size_t)(k0 + rowW) * RANK + cW * 4);
            *(float4*)&wash[rowW][cW * 4] = v;
        }
        __syncthreads();

        #pragma unroll 8
        for (int kk = 0; kk < 64; ++kk) {
            const float wv = wash[kk][r];
            acc0 += xsh[g * 4 + 0][kk] * wv;
            acc1 += xsh[g * 4 + 1][kk] * wv;
            acc2 += xsh[g * 4 + 2][kk] * wv;
            acc3 += xsh[g * 4 + 3][kk] * wv;
        }
        __syncthreads();
    }

    const int tb = t0 + g * 4;
    float accs[4] = {acc0, acc1, acc2, acc3};
    #pragma unroll
    for (int j = 0; j < 4; ++j) {
        const int t = tb + j;
        if (t < segEnd) {
            H[(size_t)t * RANK + r] = accs[j] * sc;
            if (r == 0) tokid[t] = a;
        }
    }
}

// ---------------------------------------------------------------------------
// Kernel 2: 128x128 block tile GEMM, bf16 WMMA, fused bias + LoRA epilogue.
// 8 waves: waveM in {0,1} (64 rows), waveN in {0..3} (32 cols);
// each wave: 4x2 grid of v_wmma_f32_16x16x32_bf16 accumulators.
// ---------------------------------------------------------------------------
struct FragBF { union { uint4 q[2]; v16bf v; }; };

__global__ __launch_bounds__(256) void gemm_lora_kernel(
    const float* __restrict__ x, const float* __restrict__ w,
    const float* __restrict__ bias, const float* __restrict__ wb,
    const float* __restrict__ H, const int* __restrict__ tokid,
    float* __restrict__ out)
{
    __shared__ __align__(16) unsigned short xs [2][128][72];  // 128 x 64 bf16
    __shared__ __align__(16) unsigned short wsh[2][128][72];  // 128 x 64 bf16

    const int tid   = threadIdx.x;
    const int lane  = tid & 31;
    const int wid   = tid >> 5;
    const int waveM = wid & 1;
    const int waveN = wid >> 1;
    const int l15   = lane & 15;
    const int half  = lane >> 4;
    const int mBase = blockIdx.y * 128;
    const int nBase = blockIdx.x * 128;

    const int c4 = tid & 15, r0 = tid >> 4;
    const float* xg = x + (size_t)(mBase + r0) * D_IN + c4 * 4;
    const float* wg = w + (size_t)(nBase + r0) * D_IN + c4 * 4;

    v8f acc[4][2];
    {
        v8f z;
        #pragma unroll
        for (int e = 0; e < 8; ++e) z[e] = 0.f;
        #pragma unroll
        for (int fm = 0; fm < 4; ++fm)
            #pragma unroll
            for (int fn = 0; fn < 2; ++fn) acc[fm][fn] = z;
    }

    float4 px[8], pw[8];

    auto loadT = [&](int k0) {
        #pragma unroll
        for (int i = 0; i < 8; ++i) {
            px[i] = *(const float4*)(xg + (size_t)(16 * i) * D_IN + k0);
            pw[i] = *(const float4*)(wg + (size_t)(16 * i) * D_IN + k0);
        }
    };
    auto storeT = [&](int buf) {
        #pragma unroll
        for (int i = 0; i < 8; ++i) {
            ushort4 hx = make_ushort4(f2bf(px[i].x), f2bf(px[i].y),
                                      f2bf(px[i].z), f2bf(px[i].w));
            ushort4 hw = make_ushort4(f2bf(pw[i].x), f2bf(pw[i].y),
                                      f2bf(pw[i].z), f2bf(pw[i].w));
            *(ushort4*)&xs [buf][r0 + 16 * i][c4 * 4] = hx;
            *(ushort4*)&wsh[buf][r0 + 16 * i][c4 * 4] = hw;
        }
    };
    auto computeStage = [&](int buf) {
        #pragma unroll
        for (int ks = 0; ks < 64; ks += 32) {
            FragBF af[4], bfr[2];
            // A (16x32 bf16): lane<16 -> M=lane, K = 0..7 & 16..23 (+ks)
            //                 lane>=16 -> M=lane-16, K = 8..15 & 24..31 (+ks)
            #pragma unroll
            for (int fm = 0; fm < 4; ++fm) {
                const unsigned short* p =
                    &xs[buf][waveM * 64 + fm * 16 + l15][ks + half * 8];
                af[fm].q[0] = *(const uint4*)p;
                af[fm].q[1] = *(const uint4*)(p + 16);
            }
            // B (32x16 bf16): lane<16 -> N=lane, K = 0..7 & 16..23 (+ks); lane>=16 -> K = 8..15 & 24..31
            #pragma unroll
            for (int fn = 0; fn < 2; ++fn) {
                const unsigned short* p =
                    &wsh[buf][waveN * 32 + fn * 16 + l15][ks + half * 8];
                bfr[fn].q[0] = *(const uint4*)p;
                bfr[fn].q[1] = *(const uint4*)(p + 16);
            }
            #pragma unroll
            for (int fm = 0; fm < 4; ++fm)
                #pragma unroll
                for (int fn = 0; fn < 2; ++fn)
                    acc[fm][fn] = __builtin_amdgcn_wmma_f32_16x16x32_bf16(
                        false, af[fm].v, false, bfr[fn].v,
                        (short)0, acc[fm][fn], false, false);
        }
    };

    loadT(0);
    storeT(0);
    __syncthreads();

    const int NK = D_IN / 64;  // 64 stages
    for (int kt = 0; kt < NK; ++kt) {
        const int cur = kt & 1;
        if (kt + 1 < NK) loadT((kt + 1) * 64);   // global prefetch overlaps WMMA
        computeStage(cur);
        if (kt + 1 < NK) storeT(1 - cur);        // fill other buffer
        __syncthreads();
    }

    // Epilogue: + bias + LoRA (H pre-scaled in kernel 1)
    #pragma unroll
    for (int fn = 0; fn < 2; ++fn) {
        const int n = nBase + waveN * 32 + fn * 16 + l15;
        const float bb = bias[n];
        #pragma unroll
        for (int fm = 0; fm < 4; ++fm) {
            const int mRow = mBase + waveM * 64 + fm * 16 + half * 8;
            for (int v = 0; v < 8; ++v) {
                const int m = mRow + v;
                const int a = tokid[m];
                const float* Hp  = H + (size_t)m * RANK;
                const float* wbp = wb + (size_t)a * RANK * D_OUTF + n;
                float sv = acc[fm][fn][v] + bb;
                #pragma unroll
                for (int rr = 0; rr < RANK; ++rr)
                    sv += Hp[rr] * wbp[(size_t)rr * D_OUTF];
                out[(size_t)m * D_OUTF + n] = sv;
            }
        }
    }
}

// ---------------------------------------------------------------------------
extern "C" void kernel_launch(void* const* d_in, const int* in_sizes, int n_in,
                              void* d_out, int out_size, void* d_ws, size_t ws_size,
                              hipStream_t stream)
{
    const float* x        = (const float*)d_in[0];
    const float* base_w   = (const float*)d_in[1];
    const float* base_b   = (const float*)d_in[2];
    const float* wa       = (const float*)d_in[3];
    const float* wb       = (const float*)d_in[4];
    const float* scaling  = (const float*)d_in[5];
    const int*   segment  = (const int*)d_in[6];
    const int*   lora_ids = (const int*)d_in[7];
    float*       out      = (float*)d_out;

    float* Hws   = (float*)d_ws;                                    // [T, R] f32
    int*   tokid = (int*)((char*)d_ws + (size_t)T_TOK * RANK * 4);  // [T] i32

    lora_h_kernel<<<dim3(T_TOK / 64, NADAPT), 256, 0, stream>>>(
        x, wa, scaling, segment, lora_ids, Hws, tokid);

    gemm_lora_kernel<<<dim3(D_OUTF / 128, T_TOK / 128), 256, 0, stream>>>(
        x, base_w, base_b, wb, Hws, tokid, out);
}
